// TritonLocalWindowAttn_25451976196325
// MI455X (gfx1250) — compile-verified, matrix operand order checked by reading, not run on hardware
//
#include <hip/hip_runtime.h>

// ---------------------------------------------------------------------------
// Local-window attention for MI455X (gfx1250, wave32, WMMA).
// GEMMs: bf16 WMMA (v_wmma_f32_16x16x32_bf16), fp32 accumulation, LDS-staged
// tiles via async global->LDS copies (ASYNCcnt); attention/LayerNorm is
// wave32 shuffle-reduction VALU work.
// ---------------------------------------------------------------------------

typedef __attribute__((ext_vector_type(16))) __bf16        v16bf;
typedef __attribute__((ext_vector_type(8)))  float         v8f;
typedef __attribute__((ext_vector_type(8)))  unsigned int  v8u;
typedef __attribute__((ext_vector_type(4)))  unsigned int  v4u;

// GCC-style vector type matching the async-LDS builtin's parameter type:
//   '__attribute__((__vector_size__(4 * sizeof(int)))) int *'
typedef int gv4i __attribute__((vector_size(4 * sizeof(int))));
typedef __attribute__((address_space(1))) gv4i* gas1_v4i_ptr;
typedef __attribute__((address_space(3))) gv4i* gas3_v4i_ptr;

#define E_DIM 1024
#define H_DIM 16
#define D_DIM 64
#define KWIN  17
#define LSEQ  2048
#define BATCH 4
#define MROWS (BATCH * LSEQ)   // 8192 tokens

// LDS tile geometry for the GEMM: 128 rows x 64 bf16, padded row stride of
// 72 elements (144 B, 16B-aligned so ds_load_b128 works; breaks the 2-way
// bank conflict a 128B stride would have on 64 banks).
#define KSTEP   64
#define LDS_LD  72

#if __has_builtin(__builtin_amdgcn_global_load_async_to_lds_b128) && \
    __has_builtin(__builtin_amdgcn_s_wait_asynccnt)
#define HAVE_ASYNC_LDS 1
#else
#define HAVE_ASYNC_LDS 0
#endif

__device__ __forceinline__ float bf2f(unsigned short h) {
  unsigned int u = ((unsigned int)h) << 16;
  union { unsigned int u; float f; } c; c.u = u; return c.f;
}
__device__ __forceinline__ unsigned short f2bf(float f) {
  union { float f; unsigned int u; } c; c.f = f;
  unsigned int u = c.u;
  unsigned int r = (u + 0x7FFFu + ((u >> 16) & 1u)) >> 16;  // RNE
  return (unsigned short)r;
}
__device__ __forceinline__ float siluf(float x) {
  return x * (1.0f / (1.0f + __expf(-x)));
}
__device__ __forceinline__ float wave_sum(float v) {
  #pragma unroll
  for (int off = 16; off > 0; off >>= 1) v += __shfl_xor(v, off, 32);
  return v;
}

// 16-byte global -> LDS copy; async (ASYNCcnt) when available.
__device__ __forceinline__ void copy16_g2lds(const unsigned short* g,
                                             unsigned short* l) {
#if HAVE_ASYNC_LDS
  __builtin_amdgcn_global_load_async_to_lds_b128(
      (gas1_v4i_ptr)(unsigned short*)g,   // drop const, generic -> AS1
      (gas3_v4i_ptr)l,                    // generic -> AS3
      0, 0);
#else
  *(v4u*)l = *(const v4u*)g;
#endif
}
__device__ __forceinline__ void stage_join() {
#if HAVE_ASYNC_LDS
  __builtin_amdgcn_s_wait_asynccnt(0);
#endif
  __syncthreads();
}
__device__ __forceinline__ v8u combine(v4u lo, v4u hi) {
  v8u r;
  r[0] = lo[0]; r[1] = lo[1]; r[2] = lo[2]; r[3] = lo[3];
  r[4] = hi[0]; r[5] = hi[1]; r[6] = hi[2]; r[7] = hi[3];
  return r;
}

// ---------------------------------------------------------------------------
// Elementwise fp32 -> bf16 conversion
// ---------------------------------------------------------------------------
__global__ __launch_bounds__(256) void cvt_f32_bf16(
    const float* __restrict__ in, unsigned short* __restrict__ out, size_t n) {
  size_t i = (size_t)blockIdx.x * blockDim.x + threadIdx.x;
  if (i < n) out[i] = f2bf(in[i]);
}

// W[K,N] fp32  ->  Wt[N,K] bf16  (column-major weight for contiguous-K B frags)
__global__ __launch_bounds__(256) void transpose_cvt(
    const float* __restrict__ W, unsigned short* __restrict__ Wt, int K, int N) {
  int n = blockIdx.x * 16 + threadIdx.x;
  int k = blockIdx.y * 16 + threadIdx.y;
  if (n < N && k < K) Wt[(size_t)n * K + k] = f2bf(W[(size_t)k * N + n]);
}

// ---------------------------------------------------------------------------
// bf16 WMMA GEMM with fused SiLU:  C[M,N] = silu(A[M,K] * Bt[N,K]^T)
// Workgroup: 256 threads = 8 waves; tile 128x128; wave tile 64x32 = 4x2 frags.
// K-step 64. A/B tiles cooperatively staged to LDS (once per workgroup instead
// of 4x/2x redundant per-wave global fetches), async via ASYNCcnt.
// Fragment packing follows the CDNA5 ISA 16-bit A(16x32)/B(32x16) lane layout.
// ---------------------------------------------------------------------------
template <bool OUT_F32>
__global__ __launch_bounds__(256) void gemm_bf16_silu(
    const unsigned short* __restrict__ A, const unsigned short* __restrict__ Bt,
    void* __restrict__ C, int M, int N, int K) {
  __shared__ __align__(16) unsigned short ldsA[128 * LDS_LD];
  __shared__ __align__(16) unsigned short ldsB[128 * LDS_LD];

  const int tid  = threadIdx.x;
  const int lane = tid & 31;
  const int wave = tid >> 5;              // 0..7
  const int wm   = wave >> 2;             // 0..1
  const int wn   = wave & 3;              // 0..3
  const int m0blk = blockIdx.y * 128;
  const int n0blk = blockIdx.x * 128;
  const int half = lane >> 4;             // K-half select
  const int r    = lane & 15;             // row (A) / col (B,C)

  v8f acc[4][2];
  #pragma unroll
  for (int mi = 0; mi < 4; ++mi)
    #pragma unroll
    for (int ni = 0; ni < 2; ++ni)
      #pragma unroll
      for (int v = 0; v < 8; ++v) acc[mi][ni][v] = 0.0f;

  for (int k0 = 0; k0 < K; k0 += KSTEP) {
    // ---- cooperative stage: each tile = 128 rows x 64 elems = 1024 x 16B ----
    #pragma unroll
    for (int c = 0; c < 4; ++c) {
      int chunk = tid + c * 256;          // 0..1023
      int row = chunk >> 3;               // 8 chunks (128B) per row
      int seg = chunk & 7;                // 16B segment within row
      const unsigned short* ga = A  + (size_t)(m0blk + row) * K + k0 + seg * 8;
      const unsigned short* gb = Bt + (size_t)(n0blk + row) * K + k0 + seg * 8;
      copy16_g2lds(ga, ldsA + row * LDS_LD + seg * 8);
      copy16_g2lds(gb, ldsB + row * LDS_LD + seg * 8);
      if (c == 0 && k0 + KSTEP < K) {     // warm L2 for the next K tile
        __builtin_prefetch(ga + KSTEP);
        __builtin_prefetch(gb + KSTEP);
      }
    }
    stage_join();                          // s_wait_asynccnt 0 + barrier

    // ---- two 16x16x32 WMMA sub-steps over the 64-deep LDS tile ----
    #pragma unroll
    for (int s = 0; s < 2; ++s) {
      v8u ap[4], bp[2];
      #pragma unroll
      for (int mi = 0; mi < 4; ++mi) {
        const unsigned short* p =
            ldsA + (wm * 64 + mi * 16 + r) * LDS_LD + s * 32 + 8 * half;
        ap[mi] = combine(*(const v4u*)p, *(const v4u*)(p + 16));
      }
      #pragma unroll
      for (int ni = 0; ni < 2; ++ni) {
        const unsigned short* p =
            ldsB + (wn * 32 + ni * 16 + r) * LDS_LD + s * 32 + 8 * half;
        bp[ni] = combine(*(const v4u*)p, *(const v4u*)(p + 16));
      }
      #pragma unroll
      for (int mi = 0; mi < 4; ++mi) {
        v16bf av = __builtin_bit_cast(v16bf, ap[mi]);
        #pragma unroll
        for (int ni = 0; ni < 2; ++ni) {
          v16bf bv = __builtin_bit_cast(v16bf, bp[ni]);
          acc[mi][ni] = __builtin_amdgcn_wmma_f32_16x16x32_bf16(
              false, av, false, bv, (short)0, acc[mi][ni], false, false);
        }
      }
    }
    __syncthreads();                       // protect LDS reuse next iteration
  }

  // Epilogue: C frag layout -> row = v + 8*half, col = r; fused SiLU.
  #pragma unroll
  for (int mi = 0; mi < 4; ++mi) {
    #pragma unroll
    for (int v = 0; v < 8; ++v) {
      int rowM = m0blk + wm * 64 + mi * 16 + v + 8 * half;
      #pragma unroll
      for (int ni = 0; ni < 2; ++ni) {
        float c = acc[mi][ni][v];
        float o = siluf(c);
        size_t idx = (size_t)rowM * N + (size_t)(n0blk + wn * 32 + ni * 16 + r);
        if (OUT_F32) ((float*)C)[idx] = o;
        else         ((unsigned short*)C)[idx] = f2bf(o);
      }
    }
  }
}

// ---------------------------------------------------------------------------
// Window-parameter projection: wp = silu(x @ Wwin + bwin); width/sharp per head.
// One wave (32 lanes) per token; lane j owns output column j of 2H = 32.
// ---------------------------------------------------------------------------
__global__ __launch_bounds__(32) void win_proj(
    const float* __restrict__ x, const float* __restrict__ Wwin,
    const float* __restrict__ bwin, float* __restrict__ width,
    float* __restrict__ sharp) {
  const int row = blockIdx.x;
  const int j   = threadIdx.x;            // 0..31
  const float* xr = x + (size_t)row * E_DIM;
  float acc = 0.0f;
  for (int k = 0; k < E_DIM; ++k)
    acc = fmaf(xr[k], Wwin[(size_t)k * (2 * H_DIM) + j], acc);
  float wp = siluf(acc + bwin[j]);
  float sg = 1.0f / (1.0f + __expf(-wp));
  if (j < H_DIM) width[(size_t)row * H_DIM + j] = sg * 8.0f + 0.5f;  // MAX_DIST
  else           sharp[(size_t)row * H_DIM + (j - H_DIM)] = sg * 9.5f + 0.5f;
}

// ---------------------------------------------------------------------------
// Local-window attention. One block (512 thr = 16 waves) per token; wave = head.
// Lane owns d and d+32 of D=64. Per-head LayerNorm fused via wave reductions.
// kv buffer: [MROWS, 2048] bf16, cols [0,1024) = silu(k_raw), [1024,2048) = v.
// ---------------------------------------------------------------------------
__global__ __launch_bounds__(512) void attn_window(
    const unsigned short* __restrict__ qs,   // [MROWS,1024] silu(x@Wq) bf16
    const unsigned short* __restrict__ kv,   // [MROWS,2048] silu(x@Wkv) bf16
    const float* __restrict__ q_g, const float* __restrict__ q_b,
    const float* __restrict__ k_g, const float* __restrict__ k_b,
    const float* __restrict__ width, const float* __restrict__ sharp,
    unsigned short* __restrict__ out) {      // [MROWS,1024] bf16
  const int row  = blockIdx.x;              // token index (b*L + t)
  const int h    = threadIdx.x >> 5;        // head 0..15
  const int lane = threadIdx.x & 31;
  const int t    = row & (LSEQ - 1);
  const int brow = row - t;                 // batch base row

  const size_t qbase = (size_t)row * E_DIM + h * D_DIM;
  float q0 = bf2f(qs[qbase + lane]);
  float q1 = bf2f(qs[qbase + lane + 32]);
  // q LayerNorm over D=64
  float s  = wave_sum(q0 + q1);
  float ss = wave_sum(q0 * q0 + q1 * q1);
  float mu = s * (1.0f / 64.0f);
  float rs = rsqrtf(ss * (1.0f / 64.0f) - mu * mu + 1e-5f);
  float qn0 = (q0 - mu) * rs * q_g[lane]      + q_b[lane];
  float qn1 = (q1 - mu) * rs * q_g[lane + 32] + q_b[lane + 32];

  const float wdt = width[(size_t)row * H_DIM + h];
  const float shp = sharp[(size_t)row * H_DIM + h];

  float logits[KWIN];
  #pragma unroll
  for (int w = 0; w < KWIN; ++w) {
    int src = t + w - (KWIN / 2);
    float sc = 0.0f;
    if (src >= 0 && src < LSEQ) {           // wave-uniform branch
      size_t kb = (size_t)(brow + src) * (2 * E_DIM) + h * D_DIM;
      float k0 = bf2f(kv[kb + lane]);
      float k1 = bf2f(kv[kb + lane + 32]);
      float ks  = wave_sum(k0 + k1);
      float kss = wave_sum(k0 * k0 + k1 * k1);
      float kmu = ks * (1.0f / 64.0f);
      float krs = rsqrtf(kss * (1.0f / 64.0f) - kmu * kmu + 1e-5f);
      float kn0 = (k0 - kmu) * krs * k_g[lane]      + k_b[lane];
      float kn1 = (k1 - kmu) * krs * k_g[lane + 32] + k_b[lane + 32];
      sc = wave_sum(qn0 * kn0 + qn1 * kn1) * 0.125f;   // SCALE = 64^-0.5
    }
    float rel  = fabsf((float)w - (float)(KWIN / 2));
    float soft = 1.0f / (1.0f + __expf(-(wdt - rel) * shp));
    logits[w]  = sc - (1.0f - soft) * 10000.0f;
  }

  // softmax over the 17 window slots (identical in every lane)
  float mx = logits[0];
  #pragma unroll
  for (int w = 1; w < KWIN; ++w) mx = fmaxf(mx, logits[w]);
  float denom = 0.0f;
  #pragma unroll
  for (int w = 0; w < KWIN; ++w) { logits[w] = __expf(logits[w] - mx); denom += logits[w]; }
  float inv = 1.0f / denom;

  float a0 = 0.0f, a1 = 0.0f;
  #pragma unroll
  for (int w = 0; w < KWIN; ++w) {
    int src = t + w - (KWIN / 2);
    if (src >= 0 && src < LSEQ) {
      size_t vb = (size_t)(brow + src) * (2 * E_DIM) + E_DIM + h * D_DIM;
      float p = logits[w] * inv;
      a0 = fmaf(p, bf2f(kv[vb + lane]),      a0);
      a1 = fmaf(p, bf2f(kv[vb + lane + 32]), a1);
    }
  }
  out[qbase + lane]      = f2bf(a0);
  out[qbase + lane + 32] = f2bf(a1);
}

// ---------------------------------------------------------------------------
// Host-side launch sequence
// ---------------------------------------------------------------------------
extern "C" void kernel_launch(void* const* d_in, const int* in_sizes, int n_in,
                              void* d_out, int out_size, void* d_ws, size_t ws_size,
                              hipStream_t stream) {
  const float* x    = (const float*)d_in[0];
  const float* Wq   = (const float*)d_in[1];
  const float* Wkv  = (const float*)d_in[2];
  const float* q_g  = (const float*)d_in[3];
  const float* q_b  = (const float*)d_in[4];
  const float* k_g  = (const float*)d_in[5];
  const float* k_b  = (const float*)d_in[6];
  const float* Wwin = (const float*)d_in[7];
  const float* bwin = (const float*)d_in[8];
  const float* Wout = (const float*)d_in[9];
  float* out = (float*)d_out;

  // Workspace carve-up (all 256B-aligned sizes)
  char* w = (char*)d_ws;
  unsigned short* xb    = (unsigned short*)w; w += (size_t)MROWS * E_DIM * 2;       // 16 MB
  unsigned short* WqT   = (unsigned short*)w; w += (size_t)E_DIM * E_DIM * 2;       //  2 MB
  unsigned short* WkvT  = (unsigned short*)w; w += (size_t)(2 * E_DIM) * E_DIM * 2; //  4 MB
  unsigned short* WoutT = (unsigned short*)w; w += (size_t)E_DIM * E_DIM * 2;       //  2 MB
  unsigned short* qsilu = (unsigned short*)w; w += (size_t)MROWS * E_DIM * 2;       // 16 MB
  unsigned short* kvbuf = (unsigned short*)w; w += (size_t)MROWS * 2 * E_DIM * 2;   // 32 MB
  float*          wid   = (float*)w;          w += (size_t)MROWS * H_DIM * 4;       // 0.5 MB
  float*          shp   = (float*)w;          w += (size_t)MROWS * H_DIM * 4;       // 0.5 MB
  unsigned short* aout  = (unsigned short*)w; w += (size_t)MROWS * E_DIM * 2;       // 16 MB

  // 1) Precision demotion: x -> bf16, weights -> transposed bf16 [N,K]
  {
    size_t n = (size_t)MROWS * E_DIM;
    cvt_f32_bf16<<<(unsigned)((n + 255) / 256), 256, 0, stream>>>(x, xb, n);
  }
  transpose_cvt<<<dim3(E_DIM / 16, E_DIM / 16), dim3(16, 16), 0, stream>>>(Wq,   WqT,   E_DIM, E_DIM);
  transpose_cvt<<<dim3(2 * E_DIM / 16, E_DIM / 16), dim3(16, 16), 0, stream>>>(Wkv, WkvT, E_DIM, 2 * E_DIM);
  transpose_cvt<<<dim3(E_DIM / 16, E_DIM / 16), dim3(16, 16), 0, stream>>>(Wout, WoutT, E_DIM, E_DIM);

  // 2) qsilu = silu(x @ Wq)   [8192 x 1024]
  gemm_bf16_silu<false><<<dim3(E_DIM / 128, MROWS / 128), 256, 0, stream>>>(
      xb, WqT, qsilu, MROWS, E_DIM, E_DIM);
  // 3) kv = silu(x @ Wkv)     [8192 x 2048]
  gemm_bf16_silu<false><<<dim3(2 * E_DIM / 128, MROWS / 128), 256, 0, stream>>>(
      xb, WkvT, kvbuf, MROWS, 2 * E_DIM, E_DIM);

  // 4) window parameters
  win_proj<<<MROWS, 32, 0, stream>>>(x, Wwin, bwin, wid, shp);

  // 5) fused LayerNorm + local-window attention
  attn_window<<<MROWS, 512, 0, stream>>>(qsilu, kvbuf, q_g, q_b, k_g, k_b,
                                         wid, shp, aout);

  // 6) out = silu(attn_out @ Wout)  -> fp32 d_out
  gemm_bf16_silu<true><<<dim3(E_DIM / 128, MROWS / 128), 256, 0, stream>>>(
      aout, WoutT, out, MROWS, E_DIM, E_DIM);

  (void)in_sizes; (void)n_in; (void)out_size; (void)ws_size;
}
